// GoEModel_74199855006269
// MI455X (gfx1250) — matrix operand order repaired and checked by reading, not compiled, not run on hardware
//
#include <hip/hip_runtime.h>
#include <hip/hip_bf16.h>
#include <math.h>

// ---------------------------------------------------------------- constants
#define VV 32000
#define DD 512
#define EE 4
#define NHH 8
#define FFD 2048
#define NLL 2
#define RHH 256
#define MAXVIS 2
#define BB 8
#define SS 256
#define HDD 64

typedef __attribute__((ext_vector_type(16))) __bf16 v16bf;
typedef __attribute__((ext_vector_type(8)))  __bf16 v8bf;
typedef __attribute__((ext_vector_type(8)))  float  v8f;

struct Ptr4 { const float* p[4]; };

// ---------------------------------------------------------------- TDM probe (compile-only probe,
// never launched): 6-arg clang-23 tensor_load_to_lds arity + s_wait_tensorcnt codegen.
// Uses a NULL descriptor (group0.count == 0 -> NULL tensor, architecturally a no-op).
#if defined(__gfx1250__) && __has_builtin(__builtin_amdgcn_tensor_load_to_lds) && \
    __has_builtin(__builtin_amdgcn_s_wait_tensorcnt)
typedef __attribute__((ext_vector_type(4))) unsigned int u32x4;
typedef __attribute__((ext_vector_type(8))) int          i32x8;
typedef __attribute__((ext_vector_type(4))) int          i32x4;
__global__ void tdm_probe_kernel(int* out) {
    __shared__ long long lds_sink[4];
    u32x4 g0 = {0u, 0u, 0u, 0u};          // count=0 => NULL tensor descriptor
    i32x8 g1 = {0, 0, 0, 0, 0, 0, 0, 0};
    i32x4 g2 = {0, 0, 0, 0};
    i32x4 g3 = {0, 0, 0, 0};
    i32x8 g4 = {0, 0, 0, 0, 0, 0, 0, 0};
    __builtin_amdgcn_tensor_load_to_lds(g0, g1, g2, g3, g4, 0);
    __builtin_amdgcn_s_wait_tensorcnt(0);
    if (threadIdx.x == 0) lds_sink[0] = 0;
    __syncthreads();
    if (out) out[threadIdx.x] = (int)lds_sink[threadIdx.x & 3];
}
#endif

// ---------------------------------------------------------------- reductions
__device__ __forceinline__ float block_sum(float v, float* red) {
    int t = threadIdx.x;
    red[t] = v; __syncthreads();
    for (int s = 128; s > 0; s >>= 1) {
        if (t < s) red[t] += red[t + s];
        __syncthreads();
    }
    float r = red[0]; __syncthreads();
    return r;
}

// ---------------------------------------------------------------- init state
__global__ void init_state(int* action, int* visits, int* active) {
    int t = threadIdx.x;
    if (t < BB) { action[t] = EE; active[t] = 1; }
    if (t < BB * EE) visits[t] = 0;
}

// ---------------------------------------------------------------- embed + PE + LN
__device__ __forceinline__ float pe_val(int s, int d) {
    int i = d >> 1;
    float div = __expf((float)(2 * i) * (-9.210340371976184f / (float)DD));
    float ang = (float)s * div;
    return (d & 1) ? __cosf(ang) : __sinf(ang);
}

__global__ __launch_bounds__(256)
void embed_kernel(const int* __restrict__ ids, const float* __restrict__ emb,
                  const float* __restrict__ ng, const float* __restrict__ nb,
                  float* __restrict__ repF, __bf16* __restrict__ repH) {
    __shared__ float red[256];
    int tok = blockIdx.x;          // b*256+s
    int s = tok & 255;
    int t = threadIdx.x;
    int id = ids[tok];
    float x0 = emb[(size_t)id * DD + t]       * 22.62741699796952f + pe_val(s, t);
    float x1 = emb[(size_t)id * DD + 256 + t] * 22.62741699796952f + pe_val(s, t + 256);
    float m   = block_sum(x0 + x1, red) * (1.f / DD);
    float d0 = x0 - m, d1 = x1 - m;
    float var = block_sum(d0 * d0 + d1 * d1, red) * (1.f / DD);
    float inv = rsqrtf(var + 1e-5f);
    float y0 = d0 * inv * ng[t] + nb[t];
    float y1 = d1 * inv * ng[t + 256] + nb[t + 256];
    size_t o = (size_t)tok * DD;
    repF[o + t] = y0; repF[o + 256 + t] = y1;
    repH[o + t] = (__bf16)y0; repH[o + 256 + t] = (__bf16)y1;
}

// ---------------------------------------------------------------- summary (mean over S)
__global__ __launch_bounds__(256)
void summary_kernel(const float* __restrict__ repF, float* __restrict__ summary) {
    int b = blockIdx.y;
    int d = blockIdx.x * 256 + threadIdx.x;
    const float* base = repF + (size_t)b * SS * DD + d;
    float acc = 0.f;
    for (int r = 0; r < SS; r++) acc += base[(size_t)r * DD];
    summary[b * DD + d] = acc * (1.f / SS);
}

// ---------------------------------------------------------------- router (per pass)
__global__ __launch_bounds__(256)
void router_kernel(const float* __restrict__ summary,
                   const float* __restrict__ W1, const float* __restrict__ b1,
                   const float* __restrict__ W2, const float* __restrict__ b2,
                   const float* __restrict__ lg, const float* __restrict__ lb,
                   const float* __restrict__ q,
                   int* __restrict__ action, int* __restrict__ visits, int* __restrict__ active) {
    __shared__ float sSum[DD];
    __shared__ float sH[RHH];
    __shared__ float sLg[8];
    int t = threadIdx.x;
    for (int b = 0; b < BB; b++) {
        sSum[t] = summary[b * DD + t];
        sSum[t + 256] = summary[b * DD + 256 + t];
        __syncthreads();
        const float* wr = W1 + (size_t)t * DD;
        float acc = 0.f;
        for (int k = 0; k < DD; k++) acc += wr[k] * sSum[k];
        sH[t] = fmaxf(acc + b1[t], 0.f);
        __syncthreads();
        if (t < EE + 1) {
            const float* w2r = W2 + t * RHH;
            float a2 = 0.f;
            for (int k = 0; k < RHH; k++) a2 += w2r[k] * sH[k];
            a2 += b2[t];
            sLg[t] = fminf(10.f, fmaxf(-10.f, a2));
        }
        __syncthreads();
        if (t == 0) {
            float m = 0.f, v = 0.f;
            for (int c = 0; c < EE + 1; c++) m += sLg[c];
            m *= (1.f / (EE + 1));
            for (int c = 0; c < EE + 1; c++) { float d = sLg[c] - m; v += d * d; }
            v *= (1.f / (EE + 1));
            float inv = rsqrtf(v + 1e-5f);
            float best = -3.4e38f; int bi = EE;
            for (int c = 0; c < EE + 1; c++) {
                float val = (sLg[c] - m) * inv * lg[c] + lb[c];
                if (c < EE && visits[b * EE + c] >= MAXVIS) val = -1e30f;
                val += q[c];
                if (val > best) { best = val; bi = c; }   // first-max like jnp.argmax
            }
            int sel = active[b] && (bi != EE);
            if (sel) visits[b * EE + bi] += 1;
            active[b] = sel;
            action[b] = sel ? bi : EE;
        }
        __syncthreads();
    }
}

// ---------------------------------------------------------------- generic LN (expert-selected γ/β)
__global__ __launch_bounds__(256)
void ln_kernel(const float* __restrict__ in, Ptr4 gamma, Ptr4 beta, const int* action,
               float* __restrict__ outF, __bf16* __restrict__ outH, int ldh) {
    __shared__ float red[256];
    int tok = blockIdx.x; int b = tok >> 8;
    int widx = 0;
    if (action) { int a = action[b]; if (a >= EE) return; widx = a; }
    int t = threadIdx.x;
    size_t o = (size_t)tok * DD;
    float x0 = in[o + t], x1 = in[o + 256 + t];
    float m   = block_sum(x0 + x1, red) * (1.f / DD);
    float d0 = x0 - m, d1 = x1 - m;
    float var = block_sum(d0 * d0 + d1 * d1, red) * (1.f / DD);
    float inv = rsqrtf(var + 1e-5f);
    const float* g = gamma.p[widx]; const float* be = beta.p[widx];
    float y0 = d0 * inv * g[t] + be[t];
    float y1 = d1 * inv * g[t + 256] + be[t + 256];
    if (outF) { outF[o + t] = y0; outF[o + 256 + t] = y1; }
    if (outH) {
        size_t oh = (size_t)tok * ldh;
        outH[oh + t] = (__bf16)y0; outH[oh + 256 + t] = (__bf16)y1;
    }
}

// ---------------------------------------------------------------- gated-attention combine:
// rep += LN(xn + g*a, ag, ab)
__global__ __launch_bounds__(256)
void combine_kernel(const float* __restrict__ xn, const float* __restrict__ gt,
                    const float* __restrict__ at, Ptr4 ag, Ptr4 ab, const int* action,
                    float* __restrict__ repF, __bf16* __restrict__ repH) {
    __shared__ float red[256];
    int tok = blockIdx.x; int b = tok >> 8;
    int a = action[b]; if (a >= EE) return;
    int t = threadIdx.x;
    size_t o = (size_t)tok * DD;
    float z0 = xn[o + t] + gt[o + t] * at[o + t];
    float z1 = xn[o + 256 + t] + gt[o + 256 + t] * at[o + 256 + t];
    float m   = block_sum(z0 + z1, red) * (1.f / DD);
    float d0 = z0 - m, d1 = z1 - m;
    float var = block_sum(d0 * d0 + d1 * d1, red) * (1.f / DD);
    float inv = rsqrtf(var + 1e-5f);
    const float* g = ag.p[a]; const float* be = ab.p[a];
    float r0 = repF[o + t]       + (d0 * inv * g[t] + be[t]);
    float r1 = repF[o + 256 + t] + (d1 * inv * g[t + 256] + be[t + 256]);
    repF[o + t] = r0; repF[o + 256 + t] = r1;
    repH[o + t] = (__bf16)r0; repH[o + 256 + t] = (__bf16)r1;
}

// ---------------------------------------------------------------- tag add: rep += tag[action]
__global__ __launch_bounds__(256)
void tag_kernel(Ptr4 tag, const int* action, float* __restrict__ repF, __bf16* __restrict__ repH) {
    int tok = blockIdx.x; int b = tok >> 8;
    int a = action[b]; if (a >= EE) return;
    int t = threadIdx.x;
    const float* tg = tag.p[a];
    size_t o = (size_t)tok * DD;
    float v0 = repF[o + t] + tg[t];
    float v1 = repF[o + 256 + t] + tg[256 + t];
    repF[o + t] = v0; repF[o + 256 + t] = v1;
    repH[o + t] = (__bf16)v0; repH[o + 256 + t] = (__bf16)v1;
}

// ---------------------------------------------------------------- attention (fp32, softmax)
// qkv layout per row: [q(512) | k(512) | v(512)]; one block per (s,h,b)
__global__ __launch_bounds__(256)
void attn_kernel(const float* __restrict__ qkv, const int* action, __bf16* __restrict__ oH) {
    int s = blockIdx.x, h = blockIdx.y, b = blockIdx.z;
    if (action[b] >= EE) return;
    __shared__ float sq[HDD];
    __shared__ float sp[SS];
    __shared__ float red[256];
    int t = threadIdx.x;
    size_t rowstride = 3 * DD;
    const float* bbase = qkv + (size_t)b * SS * rowstride;
    if (t < HDD) sq[t] = bbase[(size_t)s * rowstride + h * HDD + t];
    __syncthreads();
    // scores: thread t handles key row t
    const float* krow = bbase + (size_t)t * rowstride + DD + h * HDD;
    float sc = 0.f;
    #pragma unroll 8
    for (int d = 0; d < HDD; d++) sc += sq[d] * krow[d];
    sc *= 0.125f;  // 1/sqrt(64)
    // max-reduce
    red[t] = sc; __syncthreads();
    for (int st = 128; st > 0; st >>= 1) { if (t < st) red[t] = fmaxf(red[t], red[t + st]); __syncthreads(); }
    float mx = red[0]; __syncthreads();
    float e = __expf(sc - mx);
    float ssum = block_sum(e, red);
    sp[t] = e / ssum;
    __syncthreads();
    // output: thread (d, quarter) accumulates over 64 keys
    int d = t & 63, qq = t >> 6;
    const float* vbase = bbase + (size_t)(qq * 64) * rowstride + 2 * DD + h * HDD + d;
    float acc = 0.f;
    #pragma unroll 4
    for (int j = 0; j < 64; j++) acc += sp[qq * 64 + j] * vbase[(size_t)j * rowstride];
    red[t] = acc; __syncthreads();
    if (t < HDD) {
        float r = red[t] + red[t + 64] + red[t + 128] + red[t + 192];
        oH[((size_t)b * SS + s) * DD + h * HDD + t] = (__bf16)r;
    }
}

// ---------------------------------------------------------------- WMMA bf16 GEMM v2
// Y[b] = act(A_bf16[b] (256xK) @ W_sel^T + bias_sel) (+ addF).  W fp32 [N,K] row-major,
// converted to bf16 in-register.  Per-wave tile 16(M) x 64(N): 4 accumulators, A-frag
// reused 4x, 4 WMMAs per K-step.  Block = 8 waves stacked in M -> 128x64 block tile.
__global__ __launch_bounds__(256)
void wmma_gemm(const __bf16* __restrict__ A, size_t aBatch, int lda, int K,
               Ptr4 W, Ptr4 bias,
               float* __restrict__ outF, size_t ofBatch, int ldf,
               __bf16* __restrict__ outH, size_t ohBatch, int ldh,
               const float* __restrict__ addF, size_t addBatch, int ldadd,
               const int* action, int activation) {
    int b = blockIdx.z;
    int widx = 0;
    if (action) { int a = action[b]; if (a >= EE) return; widx = a; }
    const float* Wm = W.p[widx];
    const float* bp = bias.p[widx];

    int lane = threadIdx.x & 31;
    int wave = threadIdx.x >> 5;
    int m0 = blockIdx.y * 128 + wave * 16;
    int n0 = blockIdx.x * 64;
    int hf = lane >> 4;         // half-wave selector
    int lr = lane & 15;

    const __bf16* Arow = A + (size_t)b * aBatch + (size_t)(m0 + lr) * lda;
    // per-fragment W row pointers: column (n0 + 16*j + lr), K offset 8*hf
    const float* Wr0 = Wm + (size_t)(n0 +  0 + lr) * K + 8 * hf;
    const float* Wr1 = Wm + (size_t)(n0 + 16 + lr) * K + 8 * hf;
    const float* Wr2 = Wm + (size_t)(n0 + 32 + lr) * K + 8 * hf;
    const float* Wr3 = Wm + (size_t)(n0 + 48 + lr) * K + 8 * hf;
    const float* Wr[4] = {Wr0, Wr1, Wr2, Wr3};

    v8f acc[4] = {{}, {}, {}, {}};
    for (int k0 = 0; k0 < K; k0 += 32) {
        // A fragment (16x32 bf16): lanes 0-15 => K 0-7 & 16-23; lanes 16-31 => K 8-15 & 24-31
        v8bf alo = *reinterpret_cast<const v8bf*>(Arow + k0 + 8 * hf);
        v8bf ahi = *reinterpret_cast<const v8bf*>(Arow + k0 + 16 + 8 * hf);
        v16bf av;
        #pragma unroll
        for (int i = 0; i < 8; i++) { av[i] = alo[i]; av[i + 8] = ahi[i]; }
        #pragma unroll
        for (int j = 0; j < 4; j++) {
            const float4* wp = reinterpret_cast<const float4*>(Wr[j] + k0);
            // prefetch next K-step of this weight row (global_prefetch_b8)
            __builtin_prefetch((const void*)(wp + 8), 0, 1);
            float4 w0 = wp[0], w1 = wp[1], w2 = wp[4], w3 = wp[5];
            v16bf bv;
            bv[0]=(__bf16)w0.x; bv[1]=(__bf16)w0.y; bv[2]=(__bf16)w0.z; bv[3]=(__bf16)w0.w;
            bv[4]=(__bf16)w1.x; bv[5]=(__bf16)w1.y; bv[6]=(__bf16)w1.z; bv[7]=(__bf16)w1.w;
            bv[8]=(__bf16)w2.x; bv[9]=(__bf16)w2.y; bv[10]=(__bf16)w2.z; bv[11]=(__bf16)w2.w;
            bv[12]=(__bf16)w3.x; bv[13]=(__bf16)w3.y; bv[14]=(__bf16)w3.z; bv[15]=(__bf16)w3.w;
            acc[j] = __builtin_amdgcn_wmma_f32_16x16x32_bf16(false, av, false, bv,
                                                             (short)0, acc[j], false, false);
        }
    }
    // epilogue: lane writes rows m0 + hf*8 + r, column n0 + 16*j + lr
    #pragma unroll
    for (int j = 0; j < 4; j++) {
        int col = n0 + 16 * j + lr;
        float bb = bp ? bp[col] : 0.f;
        #pragma unroll
        for (int r = 0; r < 8; r++) {
            int row = m0 + hf * 8 + r;
            float v = acc[j][r] + bb;
            if (activation == 1)      v = fmaxf(v, 0.f);
            else if (activation == 2) v = 1.f / (1.f + __expf(-v));
            if (addF) v += addF[(size_t)b * addBatch + (size_t)row * ldadd + col];
            if (outF) outF[(size_t)b * ofBatch + (size_t)row * ldf + col] = v;
            if (outH) outH[(size_t)b * ohBatch + (size_t)row * ldh + col] = (__bf16)v;
        }
    }
}

// ---------------------------------------------------------------- host orchestration
extern "C" void kernel_launch(void* const* d_in, const int* in_sizes, int n_in,
                              void* d_out, int out_size, void* d_ws, size_t ws_size,
                              hipStream_t stream) {
    (void)in_sizes; (void)n_in; (void)out_size; (void)ws_size;
    // ---- flattened input indexing (setup_inputs dict order)
    const int*   ids = (const int*)d_in[0];
    const float* emb = (const float*)d_in[1];
    const float* ng  = (const float*)d_in[2];
    const float* nb  = (const float*)d_in[3];
    const float* rW1 = (const float*)d_in[4];
    const float* rb1 = (const float*)d_in[5];
    const float* rW2 = (const float*)d_in[6];
    const float* rb2 = (const float*)d_in[7];
    const float* rlg = (const float*)d_in[8];
    const float* rlb = (const float*)d_in[9];
    const float* rq  = (const float*)d_in[10];
    auto EXPW = [&](int e, int l, int w) { return (const float*)d_in[11 + e * 33 + l * 16 + w]; };
    auto TAGP = [&](int e) { return (const float*)d_in[11 + e * 33 + 32]; };
    const float* lmW = (const float*)d_in[143];
    const float* lmb = (const float*)d_in[144];

    // ---- workspace carve-up
    char* wsp = (char*)d_ws;
    size_t off = 0;
    auto carve = [&](size_t bytes) { void* p = wsp + off; off += (bytes + 255) & ~(size_t)255; return p; };
    const size_t TOK = (size_t)BB * SS;                       // 2048 tokens
    float*  repF = (float*)carve(TOK * DD * 4);
    __bf16* repH = (__bf16*)carve(TOK * DD * 2);
    float*  xnF  = (float*)carve(TOK * DD * 4);
    __bf16* catH = (__bf16*)carve(TOK * (2 * DD) * 2);        // [xn | a] bf16, ld=1024
    float*  qkvF = (float*)carve(TOK * (3 * DD) * 4);
    __bf16* oH   = (__bf16*)carve(TOK * DD * 2);
    float*  aF   = (float*)carve(TOK * DD * 4);
    float*  gF   = (float*)carve(TOK * DD * 4);
    __bf16* ynH  = (__bf16*)carve(TOK * DD * 2);
    __bf16* hffH = (__bf16*)carve(TOK * FFD * 2);
    float*  summ = (float*)carve(BB * DD * 4);
    int*    action = (int*)carve(1024);
    int*    visits = action + BB;
    int*    active = visits + BB * EE;

    init_state<<<1, 64, 0, stream>>>(action, visits, active);
    embed_kernel<<<dim3(TOK), 256, 0, stream>>>(ids, emb, ng, nb, repF, repH);

    for (int pass = 0; pass < 4; pass++) {
        summary_kernel<<<dim3(DD / 256, BB), 256, 0, stream>>>(repF, summ);
        router_kernel<<<1, 256, 0, stream>>>(summ, rW1, rb1, rW2, rb2, rlg, rlb, rq,
                                             action, visits, active);
        for (int l = 0; l < NLL; l++) {
            auto P4 = [&](int w) { Ptr4 r = {EXPW(0, l, w), EXPW(1, l, w), EXPW(2, l, w), EXPW(3, l, w)}; return r; };
            Ptr4 Wi = P4(0),  bi = P4(1),  Wo = P4(2),  bo = P4(3);
            Ptr4 gW = P4(4),  gb = P4(5),  ag = P4(6),  ab = P4(7);
            Ptr4 n1g = P4(8), n1b = P4(9), n2g = P4(10), n2b = P4(11);
            Ptr4 W1 = P4(12), b1 = P4(13), W2 = P4(14), b2 = P4(15);

            // xn = LN(rep); bf16 copy into cat[:, :512]
            ln_kernel<<<dim3(TOK), 256, 0, stream>>>(repF, n1g, n1b, action, xnF, catH, 2 * DD);
            // qkv = xn @ Wi^T + bi
            wmma_gemm<<<dim3(3 * DD / 64, 2, BB), 256, 0, stream>>>(
                catH, (size_t)SS * 2 * DD, 2 * DD, DD, Wi, bi,
                qkvF, (size_t)SS * 3 * DD, 3 * DD, nullptr, 0, 0, nullptr, 0, 0, action, 0);
            // attention -> oH (bf16)
            attn_kernel<<<dim3(SS, NHH, BB), 256, 0, stream>>>(qkvF, action, oH);
            // a = o @ Wo^T + bo ; also bf16 into cat[:, 512:]
            wmma_gemm<<<dim3(DD / 64, 2, BB), 256, 0, stream>>>(
                oH, (size_t)SS * DD, DD, DD, Wo, bo,
                aF, (size_t)SS * DD, DD, catH + DD, (size_t)SS * 2 * DD, 2 * DD,
                nullptr, 0, 0, action, 0);
            // g = sigmoid(cat(xn,a) @ gW^T + gb)
            wmma_gemm<<<dim3(DD / 64, 2, BB), 256, 0, stream>>>(
                catH, (size_t)SS * 2 * DD, 2 * DD, 2 * DD, gW, gb,
                gF, (size_t)SS * DD, DD, nullptr, 0, 0, nullptr, 0, 0, action, 2);
            // rep += LN(xn + g*a, ag, ab)
            combine_kernel<<<dim3(TOK), 256, 0, stream>>>(xnF, gF, aF, ag, ab, action, repF, repH);
            // yn = LN(rep, n2g, n2b)  (bf16 only)
            ln_kernel<<<dim3(TOK), 256, 0, stream>>>(repF, n2g, n2b, action, nullptr, ynH, DD);
            // hff = relu(yn @ W1^T + b1)  (bf16)
            wmma_gemm<<<dim3(FFD / 64, 2, BB), 256, 0, stream>>>(
                ynH, (size_t)SS * DD, DD, DD, W1, b1,
                nullptr, 0, 0, hffH, (size_t)SS * FFD, FFD, nullptr, 0, 0, action, 1);
            // rep = rep + hff @ W2^T + b2
            wmma_gemm<<<dim3(DD / 64, 2, BB), 256, 0, stream>>>(
                hffH, (size_t)SS * FFD, FFD, FFD, W2, b2,
                repF, (size_t)SS * DD, DD, repH, (size_t)SS * DD, DD,
                repF, (size_t)SS * DD, DD, action, 0);
        }
        Ptr4 tg = {TAGP(0), TAGP(1), TAGP(2), TAGP(3)};
        tag_kernel<<<dim3(TOK), 256, 0, stream>>>(tg, action, repF, repH);
    }

    // LM head: out = rep @ lmW^T + lmb  (always runs, fills entire d_out)
    Ptr4 LM = {lmW, lmW, lmW, lmW};
    Ptr4 LB = {lmb, lmb, lmb, lmb};
    wmma_gemm<<<dim3(VV / 64, 2, BB), 256, 0, stream>>>(
        repH, (size_t)SS * DD, DD, DD, LM, LB,
        (float*)d_out, (size_t)SS * VV, VV, nullptr, 0, 0, nullptr, 0, 0, nullptr, 0);
}
